// Net_75299366633924
// MI455X (gfx1250) — matrix operand order repaired and checked by reading, not compiled, not run on hardware
//
#include <hip/hip_runtime.h>

// =====================================================================
// RandLA-Net-style forward for MI455X (gfx1250, wave32, WMMA).
// GEMM: pre-padded f16 operands -> guard-free inner loop of
//       clause'd b128 loads + 4x v_wmma_f32_16x16x32_f16 + global_prefetch.
// KNN:  LDS-tiled brute force with register top-16 insertion + prefetch.
// Orchestration: stream-ordered bump arena in d_ws.
// Param leaf order: d_in[0]=x, d_in[1]=pos, then params flattened with
// sorted dict keys at every level (jax tree order).
// =====================================================================

typedef __attribute__((ext_vector_type(16))) _Float16 v16h;
typedef __attribute__((ext_vector_type(8)))  _Float16 h8;
typedef __attribute__((ext_vector_type(8)))  float    v8f;

constexpr int   CB     = 8;
constexpr int   CN     = 4096;
constexpr int   CKNN   = 16;
constexpr float CEPS   = 1e-5f;
constexpr float CSLOPE = 0.2f;

__device__ __forceinline__ float leakyf(float v) { return v >= 0.f ? v : CSLOPE * v; }

// ---------------------------------------------------------------------
// Weight prep: W[Kd x Nout] f32 row-major -> Wt[Nout x Kpad] f16, K zero-padded
// ---------------------------------------------------------------------
__global__ void w_to_f16t(const float* __restrict__ W, _Float16* __restrict__ Wt,
                          int Kd, int Nout, int Kpad) {
  int t = blockIdx.x * blockDim.x + threadIdx.x;
  if (t >= Nout * Kpad) return;
  int n = t / Kpad, k = t % Kpad;
  Wt[t] = (k < Kd) ? (_Float16)W[(size_t)k * Nout + n] : (_Float16)0.f;
}

// ---------------------------------------------------------------------
// Activation prep: A[M x Kd] f32 -> Ah[Mpad x Kpad] f16, zero-padded
// (Mpad multiple of 64, Kpad multiple of 32) -> GEMM needs no guards.
// ---------------------------------------------------------------------
__global__ void a_to_f16(const float* __restrict__ A, _Float16* __restrict__ Ah,
                         long long M, int Kd, int Kpad, long long total) {
  long long t = (long long)blockIdx.x * blockDim.x + threadIdx.x;
  if (t >= total) return;
  long long r = t / Kpad;
  int k = (int)(t % Kpad);
  Ah[t] = (r < M && k < Kd) ? (_Float16)A[r * Kd + k] : (_Float16)0.f;
}

// ---------------------------------------------------------------------
// WMMA GEMM: Y[M x Nout] = Ah[Mpad x Kpad](f16) * Wt[Nout x Kpad]^T(f16) + bias
// One wave per 64x16 output tile (4 accumulators share one B fragment).
// A frag (16x32 f16): lane l -> row m=l&15, khalf=l>>4,
//   half i -> k = i + 16*khalf  => two contiguous 16B loads
// B frag: lane l -> col n=l&15, half i -> k = i + 16*(l>>4) => one 32B load
// C/D frag: vgpr r -> M = r + 8*(l>>4), N = l&15
// ---------------------------------------------------------------------
__global__ void gemm_wmma(const _Float16* __restrict__ Ah, const _Float16* __restrict__ Wt,
                          const float* __restrict__ bias, float* __restrict__ Y,
                          int M, int Kpad, int Nout) {
  const int tm   = blockIdx.x * 64;
  const int tn   = blockIdx.y * 16;
  const int lane = threadIdx.x;
  const int half = lane >> 4;
  const int l15  = lane & 15;
  const int nc   = tn + l15;
  const bool nvalid = (nc < Nout);

  const _Float16* brow  = Wt + (size_t)(nvalid ? nc : 0) * Kpad;   // masked at store
  const _Float16* arow0 = Ah + (size_t)(tm + l15) * Kpad;
  const size_t    rstr  = (size_t)16 * Kpad;

  v8f acc[4] = {{}, {}, {}, {}};
  for (int k0 = 0; k0 < Kpad; k0 += 32) {
    // speculative prefetch of next k-step (cheap; dropped if OOB) -> global_prefetch_b8
    __builtin_prefetch(brow + k0 + 32 + 16 * half, 0, 1);
    __builtin_prefetch(arow0 + k0 + 32 + 8 * half, 0, 1);

    v16h b = *(const v16h*)(brow + k0 + 16 * half);
#pragma unroll
    for (int mt = 0; mt < 4; ++mt) {
      const _Float16* ap = arow0 + (size_t)mt * rstr + k0 + 16 * half;
      h8 a0 = *(const h8*)(ap);
      h8 a1 = *(const h8*)(ap + 8);
      v16h a = __builtin_shufflevector(a0, a1, 0, 1, 2, 3, 4, 5, 6, 7,
                                               8, 9, 10, 11, 12, 13, 14, 15);
      acc[mt] = __builtin_amdgcn_wmma_f32_16x16x32_f16(false, a, false, b,
                                                       (short)0, acc[mt], false, false);
    }
  }

  if (nvalid) {
    float bv = bias ? bias[nc] : 0.f;
#pragma unroll
    for (int mt = 0; mt < 4; ++mt) {
#pragma unroll
      for (int r = 0; r < 8; ++r) {
        int ms = tm + mt * 16 + r + 8 * half;
        if (ms < M) Y[(size_t)ms * Nout + nc] = acc[mt][r] + bv;
      }
    }
  }
}

// ---------------------------------------------------------------------
// Per-column mean/var over M rows (BN statistics). One block per column.
// ---------------------------------------------------------------------
__global__ void col_stats(const float* __restrict__ Y, int M, int Nout,
                          float* __restrict__ mean, float* __restrict__ var) {
  int c = blockIdx.x;
  float s = 0.f, s2 = 0.f;
  for (int r = threadIdx.x; r < M; r += blockDim.x) {
    float v = Y[(size_t)r * Nout + c];
    s += v; s2 += v * v;
  }
  __shared__ float ss[256], sq[256];
  ss[threadIdx.x] = s; sq[threadIdx.x] = s2;
  __syncthreads();
  for (int o = 128; o > 0; o >>= 1) {
    if ((int)threadIdx.x < o) { ss[threadIdx.x] += ss[threadIdx.x + o]; sq[threadIdx.x] += sq[threadIdx.x + o]; }
    __syncthreads();
  }
  if (threadIdx.x == 0) {
    float m = ss[0] / (float)M;
    mean[c] = m;
    var[c]  = sq[0] / (float)M - m * m;
  }
}

// ---------------------------------------------------------------------
// BN affine + optional leaky; writes into strided destination (for fused concat)
// ---------------------------------------------------------------------
__global__ void bn_act(const float* __restrict__ Y, const float* __restrict__ mean,
                       const float* __restrict__ var, const float* __restrict__ g,
                       const float* __restrict__ bt, float* __restrict__ out,
                       long long M, int Nout, int outStride, int colOff, int act) {
  long long t = (long long)blockIdx.x * blockDim.x + threadIdx.x;
  if (t >= M * Nout) return;
  long long r = t / Nout;
  int c = (int)(t % Nout);
  float v = g[c] * (Y[t] - mean[c]) * rsqrtf(var[c] + CEPS) + bt[c];
  if (act) v = leakyf(v);
  out[(size_t)r * outStride + colOff + c] = v;
}

// ---------------------------------------------------------------------
// Brute-force KNN (K=16) with 128-point LDS tiles; register insertion sort
// ---------------------------------------------------------------------
__global__ void knn_kernel(const float* __restrict__ pos, int* __restrict__ idx, int n) {
  __shared__ float sp[128 * 3];
  const int b = blockIdx.y;
  const int i = blockIdx.x * blockDim.x + threadIdx.x;
  const float* pb = pos + (size_t)b * n * 3;
  float px = 0.f, py = 0.f, pz = 0.f;
  if (i < n) { px = pb[i * 3]; py = pb[i * 3 + 1]; pz = pb[i * 3 + 2]; }
  float bd[CKNN]; int bi[CKNN];
#pragma unroll
  for (int t = 0; t < CKNN; ++t) { bd[t] = 3.0e38f; bi[t] = 0; }

  for (int j0 = 0; j0 < n; j0 += 128) {
    int jt = n - j0; if (jt > 128) jt = 128;
    for (int t = threadIdx.x; t < jt * 3; t += blockDim.x) sp[t] = pb[(size_t)j0 * 3 + t];
    // prefetch next tile while this one is consumed -> global_prefetch_b8
    __builtin_prefetch(pb + (size_t)(j0 + 128) * 3 + threadIdx.x, 0, 1);
    __syncthreads();
    for (int jj = 0; jj < jt; ++jj) {
      float dx = px - sp[jj * 3], dy = py - sp[jj * 3 + 1], dz = pz - sp[jj * 3 + 2];
      float d = dx * dx + dy * dy + dz * dz;
      if (d < bd[CKNN - 1]) {
        float cd = d; int ci = j0 + jj;
#pragma unroll
        for (int t = 0; t < CKNN; ++t) {               // predicated bubble insert
          bool sw = cd < bd[t];
          float td = sw ? bd[t] : cd;  int ti = sw ? bi[t] : ci;
          bd[t] = sw ? cd : bd[t];     bi[t] = sw ? ci : bi[t];
          cd = td; ci = ti;
        }
      }
    }
    __syncthreads();
  }
  if (i < n) {
    int* op = idx + ((size_t)b * n + i) * CKNN;
#pragma unroll
    for (int t = 0; t < CKNN; ++t) op[t] = bi[t];
  }
}

// ---------------------------------------------------------------------
// LFA front-end: gather neighbor pos/features, build rel-pos encoding rows,
// scatter x_j into first ch/2 columns of loc.
// ---------------------------------------------------------------------
__global__ void lfa_encode(const float* __restrict__ pos, const float* __restrict__ x,
                           const int* __restrict__ idx, float* __restrict__ rel,
                           float* __restrict__ loc, int n, int chH, int ch, long long total) {
  long long t = (long long)blockIdx.x * blockDim.x + threadIdx.x;
  if (t >= total) return;
  long long bi = t >> 4;
  int i = (int)(bi % n);
  int b = (int)(bi / n);
  int j = idx[t];
  const float* pb = pos + (size_t)b * n * 3;
  float pix = pb[i * 3], piy = pb[i * 3 + 1], piz = pb[i * 3 + 2];
  float pjx = pb[j * 3], pjy = pb[j * 3 + 1], pjz = pb[j * 3 + 2];
  float dx = pix - pjx, dy = piy - pjy, dz = piz - pjz;
  float dist = sqrtf(dx * dx + dy * dy + dz * dz);
  float* r = rel + (size_t)t * 10;
  r[0] = pix; r[1] = piy; r[2] = piz;
  r[3] = pjx; r[4] = pjy; r[5] = pjz;
  r[6] = dx;  r[7] = dy;  r[8] = dz;  r[9] = dist;
  const float* xj = x + ((size_t)b * n + j) * chH;
  float* l = loc + (size_t)t * ch;
  for (int c = 0; c < chH; ++c) l[c] = xj[c];
}

// ---------------------------------------------------------------------
// Fused softmax-over-K attention + weighted sum: out[row,c]
// ---------------------------------------------------------------------
__global__ void attn_agg(const float* __restrict__ att, const float* __restrict__ loc,
                         float* __restrict__ out, long long M2, int ch) {
  long long t = (long long)blockIdx.x * blockDim.x + threadIdx.x;
  if (t >= M2 * ch) return;
  long long row = t / ch;
  int c = (int)(t % ch);
  const float* ap = att + (size_t)row * CKNN * ch + c;
  const float* lp = loc + (size_t)row * CKNN * ch + c;
  float mx = -3.0e38f;
#pragma unroll
  for (int k = 0; k < CKNN; ++k) { float v = ap[(size_t)k * ch]; mx = v > mx ? v : mx; }
  float s = 0.f, acc = 0.f;
#pragma unroll
  for (int k = 0; k < CKNN; ++k) {
    float e = __expf(ap[(size_t)k * ch] - mx);
    s += e; acc += e * lp[(size_t)k * ch];
  }
  out[t] = acc / s;
}

__global__ void residual_leaky(const float* __restrict__ a, const float* __restrict__ b,
                               float* __restrict__ out, long long total) {
  long long t = (long long)blockIdx.x * blockDim.x + threadIdx.x;
  if (t < total) out[t] = leakyf(a[t] + b[t]);
}

__global__ void slice_rows(const float* __restrict__ src, float* __restrict__ dst,
                           int nsrc, int ndst, int ch, long long total) {
  long long t = (long long)blockIdx.x * blockDim.x + threadIdx.x;
  if (t >= total) return;
  int c = (int)(t % ch);
  long long r = t / ch;
  int i = (int)(r % ndst);
  int b = (int)(r / ndst);
  dst[t] = src[((size_t)b * nsrc + i) * ch + c];
}

__global__ void interp_idx_kernel(const float* __restrict__ pdst, const float* __restrict__ psrc,
                                  int* __restrict__ oi, int nd, int ns) {
  int t = blockIdx.x * blockDim.x + threadIdx.x;
  if (t >= CB * nd) return;
  int i = t % nd, b = t / nd;
  const float* pd = pdst + ((size_t)b * nd + i) * 3;
  float px = pd[0], py = pd[1], pz = pd[2];
  const float* ps = psrc + (size_t)b * ns * 3;
  float best = 3.0e38f; int bj = 0;
  for (int j = 0; j < ns; ++j) {
    float dx = px - ps[j * 3], dy = py - ps[j * 3 + 1], dz = pz - ps[j * 3 + 2];
    float d = dx * dx + dy * dy + dz * dz;
    if (d < best) { best = d; bj = j; }
  }
  oi[t] = bj;
}

__global__ void concat_up(const float* __restrict__ xc, const int* __restrict__ oi,
                          const float* __restrict__ xskip, float* __restrict__ out,
                          int nd, int nc, int ch1, int ch2, long long total) {
  long long t = (long long)blockIdx.x * blockDim.x + threadIdx.x;
  if (t >= total) return;
  int ch = ch1 + ch2;
  int c = (int)(t % ch);
  long long r = t / ch;
  int b = (int)(r / nd);
  float v;
  if (c < ch1) { int j = oi[r]; v = xc[((size_t)b * nc + j) * ch1 + c]; }
  else         { v = xskip[(size_t)r * ch2 + (c - ch1)]; }
  out[t] = v;
}

__global__ void log_softmax_k(const float* __restrict__ logits, float* __restrict__ out,
                              long long rows, int C) {
  long long t = (long long)blockIdx.x * blockDim.x + threadIdx.x;
  if (t >= rows) return;
  const float* p = logits + (size_t)t * C;
  float mx = p[0];
  for (int c = 1; c < C; ++c) mx = p[c] > mx ? p[c] : mx;
  float s = 0.f;
  for (int c = 0; c < C; ++c) s += __expf(p[c] - mx);
  float ls = __logf(s);
  float* o = out + (size_t)t * C;
  for (int c = 0; c < C; ++c) o[c] = p[c] - mx - ls;
}

// =====================================================================
// Host orchestration
// =====================================================================
namespace {

struct Arena { char* p; size_t off; };
static void* aalloc(Arena& a, size_t bytes) {
  void* r = a.p + a.off;
  a.off += (bytes + 255) & ~(size_t)255;
  return r;
}

struct SmlpP { const float *W, *b, *bt, *g; };
struct LfaP  { const float* attW; SmlpP enc, post; };
struct BlockP{ LfaP lfa1, lfa2; SmlpP mlp1, mlp2, shortcut; };

static inline unsigned gdiv(long long n, int b) { return (unsigned)((n + b - 1) / b); }

static void gemm(hipStream_t st, Arena& ar, const float* A, long long M, int Kd,
                 const float* W, const float* bias, int Nout, float* Y) {
  int Kpad = (Kd + 31) & ~31;
  long long Mpad = (M + 63) & ~63LL;
  _Float16* Wt = (_Float16*)aalloc(ar, (size_t)Nout * Kpad * sizeof(_Float16));
  int wtot = Nout * Kpad;
  w_to_f16t<<<gdiv(wtot, 256), 256, 0, st>>>(W, Wt, Kd, Nout, Kpad);
  _Float16* Ah = (_Float16*)aalloc(ar, (size_t)Mpad * Kpad * sizeof(_Float16));
  long long atot = Mpad * (long long)Kpad;
  a_to_f16<<<gdiv(atot, 256), 256, 0, st>>>(A, Ah, M, Kd, Kpad, atot);
  dim3 g(gdiv(M, 64), gdiv(Nout, 16));
  gemm_wmma<<<g, 32, 0, st>>>(Ah, Wt, bias, Y, (int)M, Kpad, Nout);
}

static void lin_apply(hipStream_t st, Arena& ar, const float* A, long long M,
                      const float* W, const float* bias, int din, int dout, float* Y) {
  size_t save = ar.off;
  gemm(st, ar, A, M, din, W, bias, dout, Y);
  ar.off = save;                       // stream-ordered reuse is safe
}

static void smlp_apply(hipStream_t st, Arena& ar, const float* A, long long M,
                       const SmlpP& p, int din, int dout,
                       float* out, int outStride, int colOff, bool act) {
  size_t save = ar.off;
  float* Y = (float*)aalloc(ar, (size_t)M * dout * sizeof(float));
  gemm(st, ar, A, M, din, p.W, p.b, dout, Y);
  float* mv = (float*)aalloc(ar, (size_t)dout * 2 * sizeof(float));
  col_stats<<<dout, 256, 0, st>>>(Y, (int)M, dout, mv, mv + dout);
  long long tot = M * dout;
  bn_act<<<gdiv(tot, 256), 256, 0, st>>>(Y, mv, mv + dout, p.g, p.bt, out,
                                         M, dout, outStride, colOff, act ? 1 : 0);
  ar.off = save;
}

static void lfa_apply(hipStream_t st, Arena& ar, const float* x, const float* posL,
                      const int* idx, int n, int ch, const LfaP& p, float* xout) {
  size_t save = ar.off;
  long long M = (long long)CB * n * CKNN;
  float* rel = (float*)aalloc(ar, (size_t)M * 10 * sizeof(float));
  float* loc = (float*)aalloc(ar, (size_t)M * ch * sizeof(float));
  lfa_encode<<<gdiv(M, 256), 256, 0, st>>>(posL, x, idx, rel, loc, n, ch / 2, ch, M);
  smlp_apply(st, ar, rel, M, p.enc, 10, ch / 2, loc, ch, ch / 2, true);
  float* att = (float*)aalloc(ar, (size_t)M * ch * sizeof(float));
  lin_apply(st, ar, loc, M, p.attW, nullptr, ch, ch, att);
  long long M2 = (long long)CB * n;
  float* agg = (float*)aalloc(ar, (size_t)M2 * ch * sizeof(float));
  attn_agg<<<gdiv(M2 * ch, 256), 256, 0, st>>>(att, loc, agg, M2, ch);
  smlp_apply(st, ar, agg, M2, p.post, ch, ch, xout, ch, 0, true);
  ar.off = save;
}

static void block_apply(hipStream_t st, Arena& ar, const float* xin, const float* posL,
                        int n, int din, int dout, const BlockP& p, float* xout) {
  size_t save = ar.off;
  int* idx = (int*)aalloc(ar, (size_t)CB * n * CKNN * sizeof(int));
  dim3 kg(gdiv(n, 128), CB);
  knn_kernel<<<kg, 128, 0, st>>>(posL, idx, n);
  long long M2 = (long long)CB * n;
  float* sc = (float*)aalloc(ar, (size_t)M2 * dout * sizeof(float));
  smlp_apply(st, ar, xin, M2, p.shortcut, din, dout, sc, dout, 0, false);
  float* h1 = (float*)aalloc(ar, (size_t)M2 * (dout / 8) * sizeof(float));
  smlp_apply(st, ar, xin, M2, p.mlp1, din, dout / 8, h1, dout / 8, 0, true);
  float* h2 = (float*)aalloc(ar, (size_t)M2 * (dout / 4) * sizeof(float));
  lfa_apply(st, ar, h1, posL, idx, n, dout / 4, p.lfa1, h2);
  float* h3 = (float*)aalloc(ar, (size_t)M2 * (dout / 2) * sizeof(float));
  lfa_apply(st, ar, h2, posL, idx, n, dout / 2, p.lfa2, h3);
  float* h4 = (float*)aalloc(ar, (size_t)M2 * dout * sizeof(float));
  smlp_apply(st, ar, h3, M2, p.mlp2, dout / 2, dout, h4, dout, 0, false);
  long long tot = M2 * dout;
  residual_leaky<<<gdiv(tot, 256), 256, 0, st>>>(h4, sc, xout, tot);
  ar.off = save;
}

static void fp_apply(hipStream_t st, Arena& ar, const float* xc, const float* pc, int nc,
                     const float* xskip, const float* pskip, int nskip,
                     int chc, int chskip, const SmlpP& p, int dout, float* out) {
  size_t save = ar.off;
  int* oi = (int*)aalloc(ar, (size_t)CB * nskip * sizeof(int));
  interp_idx_kernel<<<gdiv((long long)CB * nskip, 256), 256, 0, st>>>(pskip, pc, oi, nskip, nc);
  long long M = (long long)CB * nskip;
  int chcat = chc + chskip;
  float* cat = (float*)aalloc(ar, (size_t)M * chcat * sizeof(float));
  concat_up<<<gdiv(M * chcat, 256), 256, 0, st>>>(xc, oi, xskip, cat, nskip, nc, chc, chskip, M * chcat);
  smlp_apply(st, ar, cat, M, p, chcat, dout, out, dout, 0, true);
  ar.off = save;
}

} // namespace

extern "C" void kernel_launch(void* const* d_in, const int* in_sizes, int n_in,
                              void* d_out, int out_size, void* d_ws, size_t ws_size,
                              hipStream_t stream) {
  (void)in_sizes; (void)n_in; (void)out_size; (void)ws_size;
  const float* x   = (const float*)d_in[0];
  const float* pos = (const float*)d_in[1];

  // ---- parse params (sorted dict key order at every level) ----
  int li = 2;
  auto nx = [&]() -> const float* { return (const float*)d_in[li++]; };
  auto rdSmlp  = [&](SmlpP& s)  { s.W = nx(); s.b = nx(); s.bt = nx(); s.g = nx(); };
  auto rdLfa   = [&](LfaP& l)   { l.attW = nx(); rdSmlp(l.enc); rdSmlp(l.post); };
  auto rdBlock = [&](BlockP& b) { rdLfa(b.lfa1); rdLfa(b.lfa2);
                                  rdSmlp(b.mlp1); rdSmlp(b.mlp2); rdSmlp(b.shortcut); };
  BlockP b1, b2, b3, b4;
  rdBlock(b1); rdBlock(b2); rdBlock(b3); rdBlock(b4);
  SmlpP end1, end2; rdSmlp(end1); rdSmlp(end2);
  const float* end3W = nx(); const float* end3b = nx();
  const float* fc0W  = nx(); const float* fc0b  = nx();
  SmlpP fp1, fp2, fp3, fp4, summit;
  rdSmlp(fp1); rdSmlp(fp2); rdSmlp(fp3); rdSmlp(fp4); rdSmlp(summit);

  Arena ar{ (char*)d_ws, 0 };
  const long long MN = (long long)CB * CN;

  // fc0
  float* x0 = (float*)aalloc(ar, (size_t)MN * 32 * sizeof(float));
  lin_apply(stream, ar, x, MN, fc0W, fc0b, 3, 32, x0);

  // encoder
  float* x1 = (float*)aalloc(ar, (size_t)MN * 32 * sizeof(float));
  block_apply(stream, ar, x0, pos, CN, 32, 32, b1, x1);

  int n1 = CN / 4;
  float* x1d = (float*)aalloc(ar, (size_t)CB * n1 * 32 * sizeof(float));
  float* p1d = (float*)aalloc(ar, (size_t)CB * n1 * 3 * sizeof(float));
  slice_rows<<<gdiv((long long)CB * n1 * 32, 256), 256, 0, stream>>>(x1, x1d, CN, n1, 32, (long long)CB * n1 * 32);
  slice_rows<<<gdiv((long long)CB * n1 * 3, 256), 256, 0, stream>>>(pos, p1d, CN, n1, 3, (long long)CB * n1 * 3);
  float* x2 = (float*)aalloc(ar, (size_t)CB * n1 * 128 * sizeof(float));
  block_apply(stream, ar, x1d, p1d, n1, 32, 128, b2, x2);

  int n2 = n1 / 4;
  float* x2d = (float*)aalloc(ar, (size_t)CB * n2 * 128 * sizeof(float));
  float* p2d = (float*)aalloc(ar, (size_t)CB * n2 * 3 * sizeof(float));
  slice_rows<<<gdiv((long long)CB * n2 * 128, 256), 256, 0, stream>>>(x2, x2d, n1, n2, 128, (long long)CB * n2 * 128);
  slice_rows<<<gdiv((long long)CB * n2 * 3, 256), 256, 0, stream>>>(p1d, p2d, n1, n2, 3, (long long)CB * n2 * 3);
  float* x3 = (float*)aalloc(ar, (size_t)CB * n2 * 256 * sizeof(float));
  block_apply(stream, ar, x2d, p2d, n2, 128, 256, b3, x3);

  int n3 = n2 / 4;
  float* x3d = (float*)aalloc(ar, (size_t)CB * n3 * 256 * sizeof(float));
  float* p3d = (float*)aalloc(ar, (size_t)CB * n3 * 3 * sizeof(float));
  slice_rows<<<gdiv((long long)CB * n3 * 256, 256), 256, 0, stream>>>(x3, x3d, n2, n3, 256, (long long)CB * n3 * 256);
  slice_rows<<<gdiv((long long)CB * n3 * 3, 256), 256, 0, stream>>>(p2d, p3d, n2, n3, 3, (long long)CB * n3 * 3);
  float* x4 = (float*)aalloc(ar, (size_t)CB * n3 * 512 * sizeof(float));
  block_apply(stream, ar, x3d, p3d, n3, 256, 512, b4, x4);

  int n4 = n3 / 4;
  float* x4d = (float*)aalloc(ar, (size_t)CB * n4 * 512 * sizeof(float));
  float* p4d = (float*)aalloc(ar, (size_t)CB * n4 * 3 * sizeof(float));
  slice_rows<<<gdiv((long long)CB * n4 * 512, 256), 256, 0, stream>>>(x4, x4d, n3, n4, 512, (long long)CB * n4 * 512);
  slice_rows<<<gdiv((long long)CB * n4 * 3, 256), 256, 0, stream>>>(p3d, p4d, n3, n4, 3, (long long)CB * n4 * 3);
  float* xs = (float*)aalloc(ar, (size_t)CB * n4 * 512 * sizeof(float));
  smlp_apply(stream, ar, x4d, (long long)CB * n4, summit, 512, 512, xs, 512, 0, true);

  // decoder (feature propagation)
  float* f4 = (float*)aalloc(ar, (size_t)CB * n3 * 256 * sizeof(float));
  fp_apply(stream, ar, xs, p4d, n4, x3d, p3d, n3, 512, 256, fp4, 256, f4);
  float* f3 = (float*)aalloc(ar, (size_t)CB * n2 * 128 * sizeof(float));
  fp_apply(stream, ar, f4, p3d, n3, x2d, p2d, n2, 256, 128, fp3, 128, f3);
  float* f2 = (float*)aalloc(ar, (size_t)CB * n1 * 32 * sizeof(float));
  fp_apply(stream, ar, f3, p2d, n2, x1d, p1d, n1, 128, 32, fp2, 32, f2);
  float* f1 = (float*)aalloc(ar, (size_t)MN * 32 * sizeof(float));
  fp_apply(stream, ar, f2, p1d, n1, x1, pos, CN, 32, 32, fp1, 32, f1);

  // head
  float* h1 = (float*)aalloc(ar, (size_t)MN * 64 * sizeof(float));
  smlp_apply(stream, ar, f1, MN, end1, 32, 64, h1, 64, 0, true);
  float* h2 = (float*)aalloc(ar, (size_t)MN * 32 * sizeof(float));
  smlp_apply(stream, ar, h1, MN, end2, 64, 32, h2, 32, 0, true);
  float* logits = (float*)aalloc(ar, (size_t)MN * 13 * sizeof(float));
  lin_apply(stream, ar, h2, MN, end3W, end3b, 32, 13, logits);
  log_softmax_k<<<gdiv(MN, 256), 256, 0, stream>>>(logits, (float*)d_out, MN, 13);
}
